// ProteinMPNNFeaturesNew_14087492731567
// MI455X (gfx1250) — matrix operand order repaired and checked by reading, not compiled, not run on hardware
//
#include <hip/hip_runtime.h>
#include <hip/hip_bf16.h>
#include <math.h>

typedef __attribute__((ext_vector_type(16))) _Float16 v16h;
typedef __attribute__((ext_vector_type(8)))  _Float16 v8h;
typedef __attribute__((ext_vector_type(8)))  float    v8f;

#define B_      4
#define LRES    1024
#define TOPK    30
#define NODE_F  128
#define EDGE_F  128
#define NUM_RBF 16
#define KPAD    288          // 256 RBF + 7 orient, padded to 9*32 for WMMA K-steps
#define NKC     9            // KPAD / 32
#define EPS_DIST 1e-6f
#define EPS_LN   1e-6f
#define NEDGE   (B_*LRES*TOPK)

// ---- workspace layout (float units) ----
#define WS_ATOMS4 0                               // B*L*12  (N,Ca,C,Cb)
#define WS_OM     (WS_ATOMS4 + B_*LRES*12)        // B*L*9   local frames
#define WS_D3     (WS_OM     + B_*LRES*9)         // B*L*3   dihedral angles (padded)
#define WS_DN     (WS_D3     + B_*LRES*3)         // B*L*K   neighbor distances
#define WS_OFEAT  (WS_DN     + B_*LRES*TOPK)      // B*L*K*7 orientation features
#define WS_EIDX   (WS_OFEAT  + B_*LRES*TOPK*7)    // B*L*K   neighbor indices (int)
#define WS_WC     (WS_EIDX   + B_*LRES*TOPK)      // swizzled f16 weights: 8*9*32*16 halfs

// pair -> atom indices {N=0,Ca=1,C=2,Cb=3}; entry 0 unused (Ca-Ca from topk)
__constant__ int PA[16] = {1, 0,2,3, 1,1,1, 0,0,3, 0,2,3, 2,3,2};
__constant__ int PB[16] = {1, 0,2,3, 0,2,3, 2,3,2, 1,1,1, 0,0,3};

__device__ inline void vcross(const float* a, const float* b, float* o) {
  o[0] = a[1]*b[2] - a[2]*b[1];
  o[1] = a[2]*b[0] - a[0]*b[2];
  o[2] = a[0]*b[1] - a[1]*b[0];
}
__device__ inline void vnorm(float* a) {
  float n = sqrtf(a[0]*a[0] + a[1]*a[1] + a[2]*a[2]);
  float inv = 1.0f / fmaxf(n, 1e-12f);
  a[0] *= inv; a[1] *= inv; a[2] *= inv;
}

// ---------------- K0: repack weights into per-WMMA-fragment swizzled f16 layout ----
// element (ntile, kcI, lane, h): B-matrix 32x16 f16 lane layout:
//   col = ntile*16 + (lane&15); k = kcI*32 + ((lane<16)?0:8) + (h<8 ? h : h+8)
__global__ void k_prepw(const float* __restrict__ edist_W,
                        const float* __restrict__ eorient_W,
                        float* __restrict__ ws) {
  int t = blockIdx.x * blockDim.x + threadIdx.x;
  if (t >= 8 * NKC * 32 * 16) return;
  int h    = t & 15;
  int lane = (t >> 4) & 31;
  int kcI  = (t >> 9) % NKC;
  int nt   = t / (NKC * 512);
  int k = kcI * 32 + ((lane < 16) ? 0 : 8) + ((h < 8) ? h : h + 8);
  int n = nt * 16 + (lane & 15);
  float v = 0.0f;
  if (k < 256)      v = edist_W[n * 256 + k];
  else if (k < 263) v = eorient_W[n * 7 + (k - 256)];
  ((_Float16*)(ws + WS_WC))[t] = (_Float16)v;
}

// ---------------- K1: geometry (atoms4, frames Om, dihedral angles D3) ----------------
__global__ void k_geom(const float* __restrict__ X, float* __restrict__ ws) {
  int t = blockIdx.x * blockDim.x + threadIdx.x;
  if (t >= B_ * LRES) return;
  int b = t / LRES, l = t % LRES;
  const float* Xb = X + (long)b * LRES * 12;   // (L,4,3)

  float N[3], Ca[3], C[3], bb[3], cc[3], aa[3], Cb[3];
  #pragma unroll
  for (int c = 0; c < 3; ++c) {
    N[c]  = Xb[l*12 + 0*3 + c];
    Ca[c] = Xb[l*12 + 1*3 + c];
    C[c]  = Xb[l*12 + 2*3 + c];
    bb[c] = Ca[c] - N[c];
    cc[c] = C[c] - Ca[c];
  }
  vcross(bb, cc, aa);
  #pragma unroll
  for (int c = 0; c < 3; ++c)
    Cb[c] = -0.58273431f*aa[c] + 0.56802827f*bb[c] - 0.54067466f*cc[c] + Ca[c];
  float* A4 = ws + WS_ATOMS4 + (long)t * 12;
  #pragma unroll
  for (int c = 0; c < 3; ++c) { A4[c] = N[c]; A4[3+c] = Ca[c]; A4[6+c] = C[c]; A4[9+c] = Cb[c]; }

  float* Om = ws + WS_OM + (long)t * 9;
  if (l >= 1 && l <= LRES - 3) {
    float u2[3], u1[3], n2[3], o1[3], o3[3];
    #pragma unroll
    for (int c = 0; c < 3; ++c) {
      u2[c] = Xb[l*12 + 3 + c]     - Xb[(l-1)*12 + 3 + c];
      u1[c] = Xb[(l+1)*12 + 3 + c] - Xb[l*12 + 3 + c];
    }
    vnorm(u2); vnorm(u1);
    vcross(u2, u1, n2); vnorm(n2);
    #pragma unroll
    for (int c = 0; c < 3; ++c) o1[c] = u2[c] - u1[c];
    vnorm(o1);
    vcross(o1, n2, o3);
    #pragma unroll
    for (int c = 0; c < 3; ++c) { Om[c] = o1[c]; Om[3+c] = n2[c]; Om[6+c] = o3[c]; }
  } else {
    #pragma unroll
    for (int c = 0; c < 9; ++c) Om[c] = 0.0f;
  }

  float* D3 = ws + WS_D3 + (long)t * 3;
  #pragma unroll
  for (int c = 0; c < 3; ++c) {
    int idx = 3*l + c;
    float D = 0.0f;
    if (idx != 0 && idx < 3*LRES - 2) {
      int s = idx - 1;
      float P[4][3];
      #pragma unroll
      for (int q = 0; q < 4; ++q) {
        int tt = s + q;
        #pragma unroll
        for (int cc2 = 0; cc2 < 3; ++cc2)
          P[q][cc2] = Xb[(tt/3)*12 + (tt%3)*3 + cc2];
      }
      float u2v[3], u1v[3], u0v[3], n2v[3], n1v[3];
      #pragma unroll
      for (int cc2 = 0; cc2 < 3; ++cc2) {
        u2v[cc2] = P[1][cc2] - P[0][cc2];
        u1v[cc2] = P[2][cc2] - P[1][cc2];
        u0v[cc2] = P[3][cc2] - P[2][cc2];
      }
      vnorm(u2v); vnorm(u1v); vnorm(u0v);
      vcross(u2v, u1v, n2v); vnorm(n2v);
      vcross(u1v, u0v, n1v); vnorm(n1v);
      float cd = n2v[0]*n1v[0] + n2v[1]*n1v[1] + n2v[2]*n1v[2];
      cd = fminf(fmaxf(cd, -1.0f + 1e-7f), 1.0f - 1e-7f);
      float sg = u2v[0]*n1v[0] + u2v[1]*n1v[1] + u2v[2]*n1v[2];
      float sgn = (sg > 0.0f) ? 1.0f : ((sg < 0.0f) ? -1.0f : 0.0f);
      D = sgn * acosf(cd);
    }
    D3[c] = D;
  }
}

// ---------------- K2: per-row top-30 by iterated argmin over LDS ----------------
__global__ void k_topk(const float* __restrict__ X, const float* __restrict__ mask,
                       float* __restrict__ ws, int* __restrict__ eidx_out) {
  __shared__ float dist[LRES];
  __shared__ float rv[256];
  __shared__ int   ri[256];
  int row = blockIdx.x;                 // b*L + i
  int b = row / LRES, i = row % LRES;
  int t = threadIdx.x;
  const float* Xb = X + (long)b * LRES * 12;
  float cix = Xb[i*12+3], ciy = Xb[i*12+4], ciz = Xb[i*12+5];
  float mi = mask[row];
  for (int j = t; j < LRES; j += 256) {
    float dx = Xb[j*12+3] - cix, dy = Xb[j*12+4] - ciy, dz = Xb[j*12+5] - ciz;
    float d  = sqrtf(dx*dx + dy*dy + dz*dz + EPS_DIST);
    float m2 = mi * mask[b*LRES + j];
    dist[j] = m2 * d + (1.0f - m2) * 1e9f;
  }
  __syncthreads();
  float* Dn   = ws + WS_DN;
  int*   Eidx = (int*)(ws + WS_EIDX);
  for (int k = 0; k < TOPK; ++k) {
    float bv = 3e38f; int bi = LRES;
    #pragma unroll
    for (int s4 = 0; s4 < 4; ++s4) {
      int j = t + s4 * 256;
      float v = dist[j];
      if (v < bv || (v == bv && j < bi)) { bv = v; bi = j; }
    }
    rv[t] = bv; ri[t] = bi;
    __syncthreads();
    for (int s = 128; s > 0; s >>= 1) {
      if (t < s) {
        float ov = rv[t+s]; int oi = ri[t+s];
        if (ov < rv[t] || (ov == rv[t] && oi < ri[t])) { rv[t] = ov; ri[t] = oi; }
      }
      __syncthreads();
    }
    if (t == 0) {
      int w = ri[0];
      Eidx[row*TOPK + k]     = w;
      eidx_out[row*TOPK + k] = w;
      Dn[row*TOPK + k]       = rv[0];
      dist[w] = 3e38f;
    }
    __syncthreads();
  }
}

// ---------------- K3: per-edge orientation features (dU + quaternion) ----------------
__global__ void k_orient(float* __restrict__ ws) {
  int m = blockIdx.x * blockDim.x + threadIdx.x;
  if (m >= NEDGE) return;
  int b = m / (LRES * TOPK);
  int r = m % (LRES * TOPK);
  int i = r / TOPK;
  const int* Eidx = (const int*)(ws + WS_EIDX);
  int j = Eidx[m];
  const float* Omi = ws + WS_OM + ((long)b*LRES + i) * 9;
  const float* Omj = ws + WS_OM + ((long)b*LRES + j) * 9;
  const float* Ci  = ws + WS_ATOMS4 + ((long)b*LRES + i) * 12 + 3;
  const float* Cj  = ws + WS_ATOMS4 + ((long)b*LRES + j) * 12 + 3;
  float dX[3] = {Cj[0]-Ci[0], Cj[1]-Ci[1], Cj[2]-Ci[2]};
  float dU[3];
  #pragma unroll
  for (int a = 0; a < 3; ++a)
    dU[a] = Omi[a*3+0]*dX[0] + Omi[a*3+1]*dX[1] + Omi[a*3+2]*dX[2];
  vnorm(dU);
  float R[3][3];
  #pragma unroll
  for (int a = 0; a < 3; ++a)
    #pragma unroll
    for (int c = 0; c < 3; ++c)
      R[a][c] = Omi[0*3+a]*Omj[0*3+c] + Omi[1*3+a]*Omj[1*3+c] + Omi[2*3+a]*Omj[2*3+c];
  float xx=R[0][0], xy=R[0][1], xz=R[0][2];
  float yx=R[1][0], yy=R[1][1], yz=R[1][2];
  float zx=R[2][0], zy=R[2][1], zz=R[2][2];
  float Kq[4][4];
  Kq[0][0]=xx+yy+zz; Kq[0][1]=zy-yz;    Kq[0][2]=xz-zx;    Kq[0][3]=yx-xy;
  Kq[1][0]=zy-yz;    Kq[1][1]=xx-yy-zz; Kq[1][2]=xy+yx;    Kq[1][3]=xz+zx;
  Kq[2][0]=xz-zx;    Kq[2][1]=xy+yx;    Kq[2][2]=yy-xx-zz; Kq[2][3]=yz+zy;
  Kq[3][0]=yx-xy;    Kq[3][1]=xz+zx;    Kq[3][2]=yz+zy;    Kq[3][3]=zz-xx-yy;
  #pragma unroll
  for (int a = 0; a < 4; ++a) {
    #pragma unroll
    for (int c = 0; c < 4; ++c) Kq[a][c] *= (1.0f/3.0f);
    Kq[a][a] += 2.0f;   // shift so the top eigenvector dominates power iteration
  }
  float q[4] = {1.0f, 0.1f, 0.01f, 0.001f};
  #pragma unroll 1
  for (int it = 0; it < 24; ++it) {
    float nq[4];
    #pragma unroll
    for (int a = 0; a < 4; ++a)
      nq[a] = Kq[a][0]*q[0] + Kq[a][1]*q[1] + Kq[a][2]*q[2] + Kq[a][3]*q[3];
    float inv = __frsqrt_rn(fmaxf(nq[0]*nq[0] + nq[1]*nq[1] + nq[2]*nq[2] + nq[3]*nq[3], 1e-30f));
    #pragma unroll
    for (int a = 0; a < 4; ++a) q[a] = nq[a] * inv;
  }
  float* O = ws + WS_OFEAT + (long)m * 7;
  O[0]=dU[0]; O[1]=dU[1]; O[2]=dU[2];
  O[3]=q[0];  O[4]=q[1];  O[5]=q[2];  O[6]=q[3];
}

// ---------------- K4: fused edge features -> WMMA GEMM -> +PE/bias -> LayerNorm ----------------
__global__ void __launch_bounds__(256)
k_edge(const float* __restrict__ ws, const int* __restrict__ srel,
       const float* __restrict__ edist_b, const float* __restrict__ eorient_b,
       const float* __restrict__ gain_e,  const float* __restrict__ bias_e,
       float* __restrict__ Eout) {
  __shared__ _Float16 Ash[16 * KPAD];     // 16 edges x 288 f16 features
  __shared__ float    Eraw[16 * EDGE_F];  // pre-LN edge embeddings
  __shared__ float    red_s[16 * 16], red_q[16 * 16];
  __shared__ float    mu_s[16], rs_s[16];
  __shared__ int      jj[16], rel[16];

  int t  = threadIdx.x;
  int m0 = blockIdx.x * 16;               // NEDGE = 122880 is a multiple of 16

  for (int idx = t; idx < 16 * KPAD; idx += 256) Ash[idx] = (_Float16)0.0f;
  if (t < 16) {
    int m = m0 + t;
    int b = m / (LRES * TOPK);
    int r = m % (LRES * TOPK);
    int i = r / TOPK;
    int j = ((const int*)(ws + WS_EIDX))[m];
    jj[t]  = j;
    rel[t] = srel[((long)b*LRES + i) * LRES + j];
  }
  __syncthreads();

  // RBF features: thread = (edge e, pair p); 16 RBF bins each
  {
    int e = t >> 4, p = t & 15;
    int m = m0 + e;
    int b = m / (LRES * TOPK);
    int r = m % (LRES * TOPK);
    int i = r / TOPK;
    float d;
    if (p == 0) {
      d = ws[WS_DN + m];
    } else {
      const float* pi_ = ws + WS_ATOMS4 + ((long)b*LRES + i)    * 12 + PA[p]*3;
      const float* pj_ = ws + WS_ATOMS4 + ((long)b*LRES + jj[e])* 12 + PB[p]*3;
      float dx = pi_[0]-pj_[0], dy = pi_[1]-pj_[1], dz = pi_[2]-pj_[2];
      d = sqrtf(dx*dx + dy*dy + dz*dz + EPS_DIST);
    }
    #pragma unroll
    for (int rr = 0; rr < NUM_RBF; ++rr) {
      float mu = (20.0f / 15.0f) * rr;
      float z  = (d - mu) * (1.0f / 1.25f);
      Ash[e*KPAD + p*16 + rr] = (_Float16)__expf(-z*z);
    }
  }
  if (t < 16 * 7) {
    int e = t / 7, qf = t % 7;
    Ash[e*KPAD + 256 + qf] = (_Float16)ws[WS_OFEAT + (long)(m0 + e)*7 + qf];
  }
  __syncthreads();

  // 8 waves: wave w owns output columns [16w, 16w+16)
  int w = t >> 5, lane = t & 31;
  int arow = lane & 15;
  int akb  = (lane < 16) ? 0 : 8;    // A 16x32 f16 lane layout (two 8-half runs)
  const _Float16* Arow  = Ash + arow * KPAD + akb;
  const _Float16* Bbase = (const _Float16*)(ws + WS_WC) + ((long)w * NKC * 512) + (long)lane * 16;
  v8f acc = {};
  #pragma unroll
  for (int kcI = 0; kcI < NKC; ++kcI) {
    v8h lo = *(const v8h*)(Arow + kcI*32);        // k = base..base+7   (16B aligned)
    v8h hi = *(const v8h*)(Arow + kcI*32 + 16);   // k = base+16..+23
    v16h af = __builtin_shufflevector(lo, hi, 0,1,2,3,4,5,6,7,8,9,10,11,12,13,14,15);
    v16h bf = *(const v16h*)(Bbase + (long)kcI * 512);  // contiguous 32B fragment
    acc = __builtin_amdgcn_wmma_f32_16x16x32_f16(false, af, false, bf,
                                                 (short)0, acc, false, false);
  }

  // epilogue: + edist_b + eorient_b + positional encoding -> Eraw
  {
    int n = w*16 + (lane & 15);
    int h = n & 63;
    float embv = __expf((float)h * (-9.210340371976184f / 64.0f));
    float bsum = edist_b[n] + eorient_b[n];
    #pragma unroll
    for (int rq = 0; rq < 8; ++rq) {
      int mrow = rq + ((lane < 16) ? 0 : 8);   // C layout: lanes 16-31 hold M+8
      float ang = (float)rel[mrow] * embv;
      float pe  = (n < 64) ? __sinf(ang) : __cosf(ang);
      Eraw[mrow * EDGE_F + n] = acc[rq] + bsum + pe;
    }
  }
  __syncthreads();

  // LayerNorm (ddof=1) per edge over 128 channels
  {
    int e = t >> 4, seg = t & 15;
    float s = 0.0f, q2 = 0.0f;
    #pragma unroll
    for (int c = seg*8; c < seg*8 + 8; ++c) {
      float v = Eraw[e*EDGE_F + c];
      s += v; q2 += v*v;
    }
    red_s[e*16 + seg] = s; red_q[e*16 + seg] = q2;
  }
  __syncthreads();
  if (t < 16) {
    float s = 0.0f, q2 = 0.0f;
    #pragma unroll
    for (int seg = 0; seg < 16; ++seg) { s += red_s[t*16+seg]; q2 += red_q[t*16+seg]; }
    float mu  = s * (1.0f / 128.0f);
    float var = fmaxf((q2 - 128.0f*mu*mu) * (1.0f / 127.0f), 0.0f);
    mu_s[t] = mu;
    rs_s[t] = 1.0f / (sqrtf(var) + EPS_LN);
  }
  __syncthreads();
  {
    int e = t >> 4, seg = t & 15;
    long mg = (long)(m0 + e) * EDGE_F;
    #pragma unroll
    for (int c = seg*8; c < seg*8 + 8; ++c) {
      float v = (Eraw[e*EDGE_F + c] - mu_s[e]) * rs_s[e];
      Eout[mg + c] = gain_e[c] * v + bias_e[c];
    }
  }
}

// ---------------- K5: node dihedral features -> 42x128 MLP -> LayerNorm ----------------
__global__ void __launch_bounds__(128)
k_node(const float* __restrict__ node_W, const float* __restrict__ node_b,
       const float* __restrict__ gain_n, const float* __restrict__ bias_n,
       const float* __restrict__ ws, float* __restrict__ Vout) {
  __shared__ float feat[42];
  __shared__ float rs[128], rq[128];
  __shared__ float mu_sh, rstd_sh;
  int row = blockIdx.x;
  int b = row / LRES, l = row % LRES;
  int t = threadIdx.x;
  if (t < 21) {
    int j = t / 3, c = t % 3;
    int ls = l - 3 + j;
    float a = 0.0f;
    if (ls >= 0 && ls < LRES) a = ws[WS_D3 + ((long)b*LRES + ls)*3 + c];
    feat[t]      = __cosf(a);
    feat[21 + t] = __sinf(a);
  }
  __syncthreads();
  float acc = node_b[t];
  #pragma unroll
  for (int f = 0; f < 42; ++f) acc += feat[f] * node_W[t*42 + f];
  rs[t] = acc; rq[t] = acc * acc;
  __syncthreads();
  for (int s = 64; s > 0; s >>= 1) {
    if (t < s) { rs[t] += rs[t+s]; rq[t] += rq[t+s]; }
    __syncthreads();
  }
  if (t == 0) {
    float mu  = rs[0] * (1.0f / 128.0f);
    float var = fmaxf((rq[0] - 128.0f*mu*mu) * (1.0f / 127.0f), 0.0f);
    mu_sh = mu;
    rstd_sh = 1.0f / (sqrtf(var) + EPS_LN);
  }
  __syncthreads();
  Vout[(long)row * NODE_F + t] = gain_n[t] * (acc - mu_sh) * rstd_sh + bias_n[t];
}

extern "C" void kernel_launch(void* const* d_in, const int* in_sizes, int n_in,
                              void* d_out, int out_size, void* d_ws, size_t ws_size,
                              hipStream_t stream) {
  (void)in_sizes; (void)n_in; (void)out_size; (void)ws_size;
  const float* X         = (const float*)d_in[0];
  // d_in[1] = L (unused by the reference math)
  const float* mask      = (const float*)d_in[2];
  const int*   srel      = (const int*)  d_in[3];
  const float* node_W    = (const float*)d_in[4];
  const float* node_b    = (const float*)d_in[5];
  const float* edist_W   = (const float*)d_in[6];
  const float* edist_b   = (const float*)d_in[7];
  const float* eorient_W = (const float*)d_in[8];
  const float* eorient_b = (const float*)d_in[9];
  const float* gain_n    = (const float*)d_in[10];
  const float* bias_n    = (const float*)d_in[11];
  const float* gain_e    = (const float*)d_in[12];
  const float* bias_e    = (const float*)d_in[13];

  float* ws   = (float*)d_ws;
  float* out  = (float*)d_out;
  float* Vout = out;                                    // B*L*128
  float* Eout = out + (long)B_*LRES*NODE_F;             // B*L*K*128
  int*   Iout = (int*)(out + (long)B_*LRES*NODE_F + (long)NEDGE*EDGE_F);

  k_prepw <<<(8*NKC*512 + 255)/256, 256, 0, stream>>>(edist_W, eorient_W, ws);
  k_geom  <<<(B_*LRES + 255)/256,   256, 0, stream>>>(X, ws);
  k_topk  <<<B_*LRES,               256, 0, stream>>>(X, mask, ws, Iout);
  k_orient<<<(NEDGE + 255)/256,     256, 0, stream>>>(ws);
  k_edge  <<<NEDGE/16,              256, 0, stream>>>(ws, srel, edist_b, eorient_b,
                                                      gain_e, bias_e, Eout);
  k_node  <<<B_*LRES,               128, 0, stream>>>(node_W, node_b, gain_n, bias_n,
                                                      ws, Vout);
}
